// FeatureAdaption_81381040325419
// MI455X (gfx1250) — matrix operand order repaired and checked by reading, not compiled, not run on hardware
//
#include <hip/hip_runtime.h>
#include <hip/hip_bf16.h>
#include <math.h>

// ---------------------------------------------------------------------------
// FeatureAdaption (offset/mask conv + modulated deformable conv v2 + ReLU)
// for MI455X (gfx1250, wave32, WMMA).
//
// Both convolutions are implicit GEMMs on v_wmma_f32_16x16x32_bf16.
// K orderings are permuted (weights pre-permuted in prep kernel) so every
// 32-wide K chunk has uniform tap geometry:
//   GEMM1: K = kk*256 + c    -> (ki,kj) constant per chunk, gather = base+koff*HW
//   GEMM2: K = (g*9+k2)*64+c -> bilinear params constant per chunk
// x is pre-converted to bf16 (41 MB -> fits in 192 MB L2) so all gathers are
// 16-bit and kernel 1's fragment fill needs no conversion at all.
// ---------------------------------------------------------------------------

#define Bn   2
#define Cn   256
#define Hn   200
#define Wn   200
#define On   256
#define HWn  (Hn * Wn)
#define CM   108          // offset/mask channels (DG*3*K2)
#define MP   112          // CM padded to multiple of 16
#define KT   2304         // C * 9  == DG * Cg * 9
#define DGn  4
#define CGn  64

typedef __attribute__((ext_vector_type(16))) __bf16 v16bf;
typedef __attribute__((ext_vector_type(8)))  float  v8f;

union FragBF {
    v16bf          v;
    unsigned short u[16];
    unsigned int   d[8];
};

__device__ __forceinline__ unsigned short f2bf(float f) {
    unsigned int u = __float_as_uint(f);
    unsigned int r = u + 0x7FFFu + ((u >> 16) & 1u);   // round-to-nearest-even
    return (unsigned short)(r >> 16);
}
__device__ __forceinline__ float bf2f(unsigned short u) {
    return __uint_as_float((unsigned int)u << 16);
}

// 16-bit A-matrix 16x32 VGPR layout (contiguous, matching B):
//   lanes 0-15 row M=lane; element e <-> K=e (K 0..15)
//   lanes 16-31: K 16..31.
__device__ __forceinline__ int a_koff(int lane, int vg) {
    return (vg * 2) + ((lane & 16) ? 16 : 0);
}
// 16-bit B-matrix 32x16 layout: col N = lane%16; lanes 0-15 hold K=0..15
// (element e <-> K=e), lanes 16-31 hold K=16..31.
__device__ __forceinline__ int b_koff(int lane, int vg) {
    return (vg * 2) + ((lane & 16) ? 16 : 0);
}

__device__ __forceinline__ v8f wmma_bf16(const FragBF& a, const FragBF& b, v8f c) {
    return __builtin_amdgcn_wmma_f32_16x16x32_bf16(
        false, a.v, false, b.v, (short)0, c, false, false);
}

// ---------------------------------------------------------------------------
// Prep A: convert weights to bf16 row-major [M][KT] with permuted K.
// ---------------------------------------------------------------------------
__global__ void prep_weights(const float* __restrict__ w_om,
                             const float* __restrict__ w_dcn,
                             unsigned short* __restrict__ wom_bf,
                             unsigned short* __restrict__ wdcn_bf) {
    int i = blockIdx.x * blockDim.x + threadIdx.x;
    if (i < MP * KT) {
        int m = i / KT, knew = i - m * KT;
        int kk = knew >> 8, c = knew & 255;
        float v = (m < CM) ? w_om[m * KT + c * 9 + kk] : 0.0f;
        wom_bf[i] = f2bf(v);
    }
    if (i < On * KT) {
        int m = i / KT, knew = i - m * KT;
        int gk = knew >> 6, c = knew & 63;
        int g = gk / 9, k2 = gk - g * 9;
        wdcn_bf[i] = f2bf(w_dcn[m * KT + (g * CGn + c) * 9 + k2]);
    }
}

// ---------------------------------------------------------------------------
// Prep B: x f32 -> bf16, vectorized float4 -> 2x packed bf16 pairs.
// ---------------------------------------------------------------------------
__global__ void prep_x(const float* __restrict__ x,
                       unsigned short* __restrict__ xbf, int total4) {
    int i = blockIdx.x * blockDim.x + threadIdx.x;
    if (i < total4) {
        float4 v = ((const float4*)x)[i];
        unsigned int lo = (unsigned int)f2bf(v.x) | ((unsigned int)f2bf(v.y) << 16);
        unsigned int hi = (unsigned int)f2bf(v.z) | ((unsigned int)f2bf(v.w) << 16);
        ((uint2*)xbf)[i] = make_uint2(lo, hi);
    }
}

// ---------------------------------------------------------------------------
// Kernel 1: offset/mask 3x3 conv as implicit GEMM. One wave per block,
// 32 spatial positions (two N-tiles), 7 M-tiles. K chunk = one tap, 32
// channels: im2col gathers are raw u16 loads at base + koff*HW (immediate
// offsets), bf16 bits land directly in the WMMA fragment.
// Epilogue: +bias; sigmoid on mask channels (72..107).
// ---------------------------------------------------------------------------
__global__ __launch_bounds__(32)
void om_conv_kernel(const unsigned short* __restrict__ xbf,
                    const unsigned short* __restrict__ wom_bf,
                    const float* __restrict__ bias,
                    float* __restrict__ om) {
    const int lane = threadIdx.x & 31;
    const int tile = blockIdx.x;        // 0..1249
    const int bb   = blockIdx.y;
    const int n    = lane & 15;
    const int hi16 = (lane & 16) ? 16 : 0;

    const int pos0 = tile * 32 + n;
    const int pos1 = pos0 + 16;
    const int h0 = pos0 / Wn, w0 = pos0 - h0 * Wn;
    const int h1 = pos1 / Wn, w1 = pos1 - h1 * Wn;

    v8f acc[14];
    v8f zero = {0.f, 0.f, 0.f, 0.f, 0.f, 0.f, 0.f, 0.f};
    #pragma unroll
    for (int t = 0; t < 14; ++t) acc[t] = zero;

    const unsigned short* xb = xbf + (size_t)bb * Cn * HWn;

    for (int kb = 0; kb < KT; kb += 32) {
        const int kk = kb >> 8;          // tap index, constant over chunk
        const int c0 = kb & 255;         // channel base
        const int ki = kk / 3, kj = kk - ki * 3;

        FragBF b0, b1;
        #pragma unroll
        for (int vg = 0; vg < 8; ++vg) { b0.d[vg] = 0u; b1.d[vg] = 0u; }
        {
            int y = h0 - 1 + ki, xx = w0 - 1 + kj;
            if (y >= 0 && y < Hn && xx >= 0 && xx < Wn) {
                const unsigned short* base = xb + (size_t)c0 * HWn + y * Wn + xx;
                #pragma unroll
                for (int e = 0; e < 16; ++e)
                    b0.u[e] = base[(e + hi16) * HWn];
            }
        }
        {
            int y = h1 - 1 + ki, xx = w1 - 1 + kj;
            if (y >= 0 && y < Hn && xx >= 0 && xx < Wn) {
                const unsigned short* base = xb + (size_t)c0 * HWn + y * Wn + xx;
                #pragma unroll
                for (int e = 0; e < 16; ++e)
                    b1.u[e] = base[(e + hi16) * HWn];
            }
        }
        #pragma unroll
        for (int mt = 0; mt < 7; ++mt) {
            const unsigned short* arow = wom_bf + (size_t)(mt * 16 + n) * KT + kb;
            FragBF af;
            #pragma unroll
            for (int vg = 0; vg < 8; ++vg)
                af.d[vg] = *(const unsigned int*)(arow + a_koff(lane, vg));
            acc[mt]     = wmma_bf16(af, b0, acc[mt]);
            acc[7 + mt] = wmma_bf16(af, b1, acc[7 + mt]);
        }
    }

    #pragma unroll
    for (int nt = 0; nt < 2; ++nt) {
        int pos = tile * 32 + n + nt * 16;
        #pragma unroll
        for (int mt = 0; mt < 7; ++mt) {
            #pragma unroll
            for (int r = 0; r < 8; ++r) {
                int m = mt * 16 + r + ((lane & 16) ? 8 : 0);
                if (m < CM) {
                    float v = acc[nt * 7 + mt][r] + bias[m];
                    if (m >= 72) v = 1.0f / (1.0f + __expf(-v));  // mask sigmoid
                    om[((size_t)bb * CM + m) * HWn + pos] = v;
                }
            }
        }
    }
}

// ---------------------------------------------------------------------------
// Kernel 2: modulated deformable conv as implicit GEMM [256 x 2304] x B.
// 128 threads (4 waves), 32 positions/block. Bilinear params precomputed to
// LDS once (36 combos x 32 positions); sampled 32x32 bf16 B tile staged into
// a DOUBLE-BUFFERED LDS tile (one barrier per K chunk: gathers for chunk i+1
// overlap the 8 WMMAs of chunk i). Each wave: 4 M-tiles x 2 N-tiles.
// ---------------------------------------------------------------------------
__global__ __launch_bounds__(128)
void dcn_kernel(const unsigned short* __restrict__ xbf,
                const unsigned short* __restrict__ wdcn_bf,
                const float* __restrict__ om,
                float* __restrict__ out) {
    __shared__ int   sY0[1152], sX0[1152];
    __shared__ float sWY[1152], sWX[1152], sM[1152];
    __shared__ __align__(8) unsigned short sB[2][32][34];   // [buf][n][k]

    const int tid  = threadIdx.x;
    const int lane = tid & 31;
    const int wave = tid >> 5;
    const int tile = blockIdx.x;       // 0..1249
    const int bb   = blockIdx.y;

    // ---- bilinear parameters: 36 (g,k2) combos x 32 positions ----
    for (int i = tid; i < 1152; i += 128) {
        int n  = i & 31;
        int gk = i >> 5;               // g*9 + k2
        int g  = gk / 9, k2 = gk - g * 9;
        int pos = tile * 32 + n;
        int h = pos / Wn, w = pos - (pos / Wn) * Wn;
        const float* omb = om + (size_t)bb * CM * HWn;
        float dy = omb[(size_t)(g * 18 + k2 * 2 + 0) * HWn + pos];
        float dx = omb[(size_t)(g * 18 + k2 * 2 + 1) * HWn + pos];
        float mm = omb[(size_t)(72 + g * 9 + k2) * HWn + pos];  // sigmoided
        int ki = k2 / 3, kj = k2 - ki * 3;
        float py = (float)(h - 1 + ki) + dy;
        float px = (float)(w - 1 + kj) + dx;
        float y0f = floorf(py), x0f = floorf(px);
        sY0[i] = (int)y0f; sX0[i] = (int)x0f;
        sWY[i] = py - y0f; sWX[i] = px - x0f; sM[i] = mm;
    }

    const unsigned short* xb = xbf + (size_t)bb * Cn * HWn;

    // Stage one 32K x 32N sampled/modulated bf16 tile into sB[buf].
    auto stage = [&](int kb, int buf) {
        const int gk = kb >> 6;        // (g*9+k2), constant over chunk
        const int c0 = kb & 63;        // 0 or 32
        const int g  = gk / 9;
        const int chbase = g * CGn + c0;
        for (int s = tid; s < 1024; s += 128) {
            int kl = s >> 5, n = s & 31;
            int pi = gk * 32 + n;
            int y0 = sY0[pi], x0 = sX0[pi];
            float wy = sWY[pi], wx = sWX[pi], mm = sM[pi];
            const unsigned short* xc = xb + (size_t)(chbase + kl) * HWn;
            int y1 = y0 + 1, x1 = x0 + 1;
            bool yv0 = (y0 >= 0) & (y0 < Hn), yv1 = (y1 >= 0) & (y1 < Hn);
            bool xv0 = (x0 >= 0) & (x0 < Wn), xv1 = (x1 >= 0) & (x1 < Wn);
            float v00 = (yv0 & xv0) ? bf2f(xc[y0 * Wn + x0]) : 0.f;
            float v01 = (yv0 & xv1) ? bf2f(xc[y0 * Wn + x1]) : 0.f;
            float v10 = (yv1 & xv0) ? bf2f(xc[y1 * Wn + x0]) : 0.f;
            float v11 = (yv1 & xv1) ? bf2f(xc[y1 * Wn + x1]) : 0.f;
            float v = (1.f - wy) * ((1.f - wx) * v00 + wx * v01)
                    +         wy * ((1.f - wx) * v10 + wx * v11);
            sB[buf][n][kl] = f2bf(v * mm);
        }
    };

    __syncthreads();                   // params ready
    stage(0, 0);
    __syncthreads();                   // first tile ready

    v8f acc[8];
    v8f zero = {0.f, 0.f, 0.f, 0.f, 0.f, 0.f, 0.f, 0.f};
    #pragma unroll
    for (int t = 0; t < 8; ++t) acc[t] = zero;

    int cur = 0;
    for (int kb = 0; kb < KT; kb += 32) {
        // ---- consume: two B fragments from sB[cur] ----
        const int n = lane & 15;
        FragBF bf0, bf1;
        #pragma unroll
        for (int vg = 0; vg < 8; ++vg) {
            bf0.d[vg] = *(const unsigned int*)(&sB[cur][n][b_koff(lane, vg)]);
            bf1.d[vg] = *(const unsigned int*)(&sB[cur][n + 16][b_koff(lane, vg)]);
        }

        // ---- prefetch next chunk into the other buffer (overlaps WMMAs) ----
        if (kb + 32 < KT) stage(kb + 32, cur ^ 1);

        #pragma unroll
        for (int t = 0; t < 4; ++t) {
            int m = (wave * 4 + t) * 16 + n;
            const unsigned short* arow = wdcn_bf + (size_t)m * KT + kb;
            FragBF af;
            #pragma unroll
            for (int vg = 0; vg < 8; ++vg)
                af.d[vg] = *(const unsigned int*)(arow + a_koff(lane, vg));
            acc[t * 2 + 0] = wmma_bf16(af, bf0, acc[t * 2 + 0]);
            acc[t * 2 + 1] = wmma_bf16(af, bf1, acc[t * 2 + 1]);
        }
        __syncthreads();               // next tile ready / this tile free
        cur ^= 1;
    }

    // ---- epilogue: ReLU, store ----
    const int n = lane & 15;
    #pragma unroll
    for (int t = 0; t < 4; ++t) {
        #pragma unroll
        for (int nt = 0; nt < 2; ++nt) {
            int pos = tile * 32 + n + nt * 16;
            #pragma unroll
            for (int r = 0; r < 8; ++r) {
                int m = (wave * 4 + t) * 16 + r + ((lane & 16) ? 8 : 0);
                float v = acc[t * 2 + nt][r];
                out[((size_t)bb * On + m) * HWn + pos] = v > 0.f ? v : 0.f;
            }
        }
    }
}

// ---------------------------------------------------------------------------
// Workspace layout (bytes):
//   [0,          516096)          wom_bf  : 112*2304 bf16 (K-permuted)
//   [516096,     1695744)         wdcn_bf : 256*2304 bf16 (K-permuted)
//   [1695744,    36255744)        om      : 2*108*40000 f32
//   [36255744,   77215744)        xbf     : 2*256*40000 bf16
// ---------------------------------------------------------------------------
extern "C" void kernel_launch(void* const* d_in, const int* in_sizes, int n_in,
                              void* d_out, int out_size, void* d_ws, size_t ws_size,
                              hipStream_t stream) {
    const float* x     = (const float*)d_in[0];
    const float* w_om  = (const float*)d_in[1];
    const float* b_om  = (const float*)d_in[2];
    const float* w_dcn = (const float*)d_in[3];
    float* out = (float*)d_out;

    char* ws = (char*)d_ws;
    unsigned short* wom_bf  = (unsigned short*)(ws);
    unsigned short* wdcn_bf = (unsigned short*)(ws + 516096);
    float*          om      = (float*)(ws + 1695744);
    unsigned short* xbf     = (unsigned short*)(ws + 36255744);

    {
        int total = On * KT;   // >= MP*KT
        prep_weights<<<(total + 255) / 256, 256, 0, stream>>>(w_om, w_dcn, wom_bf, wdcn_bf);
    }
    {
        int total4 = (Bn * Cn * HWn) / 4;
        prep_x<<<(total4 + 255) / 256, 256, 0, stream>>>(x, xbf, total4);
    }

    dim3 grid(HWn / 32, Bn);
    om_conv_kernel<<<grid, 32, 0, stream>>>(xbf, wom_bf, b_om, om);
    dcn_kernel<<<grid, 128, 0, stream>>>(xbf, wdcn_bf, om, out);
}
